// ResidualNet_50079318671639
// MI455X (gfx1250) — matrix operand order repaired and checked by reading, not compiled, not run on hardware
//
#include <hip/hip_runtime.h>
#include <hip/hip_bf16.h>

typedef _Float16 v16h __attribute__((ext_vector_type(16)));
typedef _Float16 v8h  __attribute__((ext_vector_type(8)));
typedef float    v8f  __attribute__((ext_vector_type(8)));

#define HG_LEVELS 4
#define HG_LOG2_T 19
#define HG_T (1u << HG_LOG2_T)
#define HG_NMIN 200.0f
#define LDS_STRIDE 72   // halves; 144B row stride -> conflict-free b128 A-frag loads

// leaky_relu(x) with slope 0.01, exact algebraic form: 0.505*x + 0.495*|x|
// -> v_mul + v_fma (|x| is a free src modifier), no NaN-canonicalize max.
__device__ __forceinline__ float leaky(float x) {
    return fmaf(0.495f, fabsf(x), 0.505f * x);
}

// single-op XOR lane shuffle: ds_swizzle_b32, group-of-32 mode,
// offset = {xor_mask[14:10], or_mask=0, and_mask=0x1f}
template <int XM>
__device__ __forceinline__ int swz_xor_i(int x) {
    return __builtin_amdgcn_ds_swizzle(x, (XM << 10) | 0x1f);
}
template <int XM>
__device__ __forceinline__ float swz_xor_f(float x) {
    return __int_as_float(__builtin_amdgcn_ds_swizzle(__float_as_int(x), (XM << 10) | 0x1f));
}

__device__ __forceinline__ unsigned pack2(float a, float b) {
    union { _Float16 h[2]; unsigned u; } cv;
    cv.h[0] = (_Float16)a; cv.h[1] = (_Float16)b; return cv.u;
}

// B fragment (32x16 tile, 16-bit): lanes 0-15 hold K=kBase..kBase+15 of column n,
// lanes 16-31 hold the next 16 K values (caller passes kBase = chunk + 16*half).
__device__ __forceinline__ v16h load_bfrag(const float* __restrict__ W, int kBase, int kMax, int n) {
    v16h b;
#pragma unroll
    for (int i = 0; i < 16; ++i) {
        int k = kBase + i;
        float v = (k < kMax) ? W[k * 64 + n] : 0.0f;
        b[i] = (_Float16)v;
    }
    return b;
}

__device__ __forceinline__ void encode_level(const float* __restrict__ tab,
                                             float x01, float y01, float z01,
                                             float res, float& f0, float& f1) {
    float px = x01 * res, py = y01 * res, pz = z01 * res;
    float fx = floorf(px), fy = floorf(py), fz = floorf(pz);
    float tx = px - fx, ty = py - fy, tz = pz - fz;
    unsigned bx = (unsigned)(int)fx, by = (unsigned)(int)fy, bz = (unsigned)(int)fz;
    // per-axis hash terms / weights; corner loop just XORs + FMAs
    unsigned hyv[2] = {by * 2654435761u, (by + 1u) * 2654435761u};
    unsigned hzv[2] = {bz * 805459861u,  (bz + 1u) * 805459861u};
    float wx[2] = {1.0f - tx, tx};
    float wy[2] = {1.0f - ty, ty};
    float wz[2] = {1.0f - tz, tz};
    float a0 = 0.0f, a1 = 0.0f;
#pragma unroll
    for (int c = 0; c < 8; ++c) {
        unsigned cx = bx + (c & 1);
        unsigned h = cx ^ hyv[(c >> 1) & 1] ^ hzv[(c >> 2) & 1];
        unsigned idx = h & (HG_T - 1u);
        const float2 f = ((const float2*)tab)[idx];   // global_load_b64 gather (L2-resident)
        float w = wx[c & 1] * wy[(c >> 1) & 1] * wz[(c >> 2) & 1];
        a0 = fmaf(w, f.x, a0);
        a1 = fmaf(w, f.y, a1);
    }
    f0 = a0; f1 = a1;
}

__global__ __launch_bounds__(256, 1)
void ngp_sdf_wmma_kernel(const float* __restrict__ xyz,
                         const float* __restrict__ tables,
                         const float* __restrict__ W_in,
                         const float* __restrict__ W_h,
                         const float* __restrict__ W_out,
                         float* __restrict__ out, int N) {
    __shared__ _Float16 stage[8][16 * LDS_STRIDE];   // per-wave 16x64 f16 transpose tile

    const int lane   = threadIdx.x & 31;
    const int waveId = threadIdx.x >> 5;
    const int half   = lane >> 4;     // 0: lanes 0-15, 1: lanes 16-31
    const int l16    = lane & 15;

    // ---- per-wave weight fragments in registers (loaded once) ----
    v16h bIn[4];
#pragma unroll
    for (int t = 0; t < 4; ++t)
        bIn[t] = load_bfrag(W_in, half * 16, 8, l16 + 16 * t);   // K padded 8->32

    v16h bH[2][4];
#pragma unroll
    for (int kc = 0; kc < 2; ++kc)
#pragma unroll
        for (int t = 0; t < 4; ++t)
            bH[kc][t] = load_bfrag(W_h, kc * 32 + half * 16, 64, l16 + 16 * t);

    float wOut[4];
#pragma unroll
    for (int t = 0; t < 4; ++t) wOut[t] = W_out[l16 + 16 * t];

    const int numTiles = (N + 15) >> 4;
    const int wavesPerBlock = blockDim.x >> 5;
    const int tileStep = gridDim.x * wavesPerBlock;
    _Float16* my = &stage[waveId][0];

    for (int tileBase = blockIdx.x * wavesPerBlock; tileBase < numTiles;
         tileBase += tileStep) {
        const int tile = tileBase + waveId;
        const bool active = (tile < numTiles);     // physically wave-uniform
        const int m0 = tile << 4;

        if (active) {
            // prefetch next tile's xyz (global_prefetch_b8) while gathers run
            {
                int pn = (tile + tileStep) << 4;
                if (pn + l16 < N)
                    __builtin_prefetch(xyz + 3 * (pn + l16), 0, 3);
            }

            // ---- hash-grid encode: lanes L, L+16 share point L, split levels ----
            int p = m0 + l16;
            if (p >= N) p = N - 1;
            const float x01 = (xyz[3 * p + 0] + 5.0f) * 0.1f;
            const float y01 = (xyz[3 * p + 1] + 5.0f) * 0.1f;
            const float z01 = (xyz[3 * p + 2] + 5.0f) * 0.1f;

            float f[4];
#pragma unroll
            for (int j = 0; j < 2; ++j) {
                const int l = half * 2 + j;
                const float res = HG_NMIN * (float)(1 << l);
                encode_level(tables + (size_t)l * HG_T * 2, x01, y01, z01, res,
                             f[2 * j], f[2 * j + 1]);
            }
            unsigned own0 = pack2(f[0], f[1]);
            unsigned own1 = pack2(f[2], f[3]);
            unsigned oth0 = (unsigned)swz_xor_i<16>((int)own0);   // single ds_swizzle
            unsigned oth1 = (unsigned)swz_xor_i<16>((int)own1);

            // ---- layer-1 A fragment: 16x32, K=0..7 real, rest zero padding ----
            v16h a1 = {};
            if (half == 0) {
                union { unsigned u; _Float16 h[2]; } c0, c1, c2, c3;
                c0.u = own0; c1.u = own1; c2.u = oth0; c3.u = oth1;
                a1[0] = c0.h[0]; a1[1] = c0.h[1];
                a1[2] = c1.h[0]; a1[3] = c1.h[1];
                a1[4] = c2.h[0]; a1[5] = c2.h[1];
                a1[6] = c3.h[0]; a1[7] = c3.h[1];
            }

            // ---- layer 1: enc[16,32pad] @ W_in[32,64] -> h1[16,64] (4 WMMAs) ----
            v8f h1[4];
#pragma unroll
            for (int t = 0; t < 4; ++t) {
                v8f z = {};
                h1[t] = __builtin_amdgcn_wmma_f32_16x16x32_f16(
                    false, a1, false, bIn[t], (short)0, z, false, false);
            }
            // leaky-ReLU + write f16 tile to LDS (C-layout -> memory tile)
#pragma unroll
            for (int t = 0; t < 4; ++t)
#pragma unroll
                for (int r = 0; r < 8; ++r)
                    my[(r + 8 * half) * LDS_STRIDE + (l16 + 16 * t)] =
                        (_Float16)leaky(h1[t][r]);
        }
        __syncthreads();
        if (active) {
            // ---- layer-2 A fragments from LDS (transpose realized by layout) ----
            v16h a2[2];
#pragma unroll
            for (int kc = 0; kc < 2; ++kc) {
                const v8h lo = *(const v8h*)(my + l16 * LDS_STRIDE + kc * 32 + half * 16);
                const v8h hi = *(const v8h*)(my + l16 * LDS_STRIDE + kc * 32 + half * 16 + 8);
#pragma unroll
                for (int i = 0; i < 8; ++i) { a2[kc][i] = lo[i]; a2[kc][8 + i] = hi[i]; }
            }

            // ---- layer 2: h1[16,64] @ W_h[64,64] (8 WMMAs, C-chained) ----
            v8f acc2[4];
#pragma unroll
            for (int t = 0; t < 4; ++t) {
                v8f z = {};
                v8f c = __builtin_amdgcn_wmma_f32_16x16x32_f16(
                    false, a2[0], false, bH[0][t], (short)0, z, false, false);
                acc2[t] = __builtin_amdgcn_wmma_f32_16x16x32_f16(
                    false, a2[1], false, bH[1][t], (short)0, c, false, false);
            }

            // ---- layer 3: leaky + dot with W_out + 16-lane butterfly reduce ----
            float part[8];
#pragma unroll
            for (int r = 0; r < 8; ++r) {
                float s = 0.0f;
#pragma unroll
                for (int t = 0; t < 4; ++t)
                    s = fmaf(leaky(acc2[t][r]), wOut[t], s);
                part[r] = s;
            }
            // XOR butterfly over the 16-lane halves via single-op ds_swizzle
#pragma unroll
            for (int r = 0; r < 8; ++r) part[r] += swz_xor_f<1>(part[r]);
#pragma unroll
            for (int r = 0; r < 8; ++r) part[r] += swz_xor_f<2>(part[r]);
#pragma unroll
            for (int r = 0; r < 8; ++r) part[r] += swz_xor_f<4>(part[r]);
#pragma unroll
            for (int r = 0; r < 8; ++r) part[r] += swz_xor_f<8>(part[r]);

            if (l16 == 0) {
                const int rowBase = m0 + 8 * half;
                if (rowBase + 8 <= N) {
                    float4 o0 = make_float4(part[0] * 0.1f, part[1] * 0.1f,
                                            part[2] * 0.1f, part[3] * 0.1f);
                    float4 o1 = make_float4(part[4] * 0.1f, part[5] * 0.1f,
                                            part[6] * 0.1f, part[7] * 0.1f);
                    *(float4*)(out + rowBase)     = o0;   // global_store_b128
                    *(float4*)(out + rowBase + 4) = o1;
                } else {
#pragma unroll
                    for (int r = 0; r < 8; ++r) {
                        int row = rowBase + r;
                        if (row < N) out[row] = part[r] * 0.1f;
                    }
                }
            }
        }
        __syncthreads();
    }
}

extern "C" void kernel_launch(void* const* d_in, const int* in_sizes, int n_in,
                              void* d_out, int out_size, void* d_ws, size_t ws_size,
                              hipStream_t stream) {
    const float* xyz    = (const float*)d_in[0];
    const float* tables = (const float*)d_in[1];
    const float* W_in   = (const float*)d_in[2];
    const float* W_h    = (const float*)d_in[3];
    const float* W_out  = (const float*)d_in[4];
    float* out = (float*)d_out;

    const int N = in_sizes[0] / 3;
    const int numTiles = (N + 15) / 16;
    const int wavesPerBlock = 8;                 // 256 threads = 8 wave32
    int blocksNeeded = (numTiles + wavesPerBlock - 1) / wavesPerBlock;
    int grid = blocksNeeded < 4096 ? blocksNeeded : 4096;
    if (grid < 1) grid = 1;

    ngp_sdf_wmma_kernel<<<grid, 256, 0, stream>>>(xyz, tables, W_in, W_h, W_out, out, N);
}
